// Attention_42460046688842
// MI455X (gfx1250) — matrix operand order repaired and checked by reading, not compiled, not run on hardware
//
#include <hip/hip_runtime.h>

#define DD   768
#define SS   2048
#define NB   2
#define NH   12
#define HDIM 64
#define MTOT (NB * SS)   // 4096

typedef __bf16 bf16;
typedef __attribute__((ext_vector_type(4)))  __bf16 v4bf;
typedef __attribute__((ext_vector_type(8)))  __bf16 v8bf;
typedef __attribute__((ext_vector_type(16))) __bf16 v16bf;
typedef __attribute__((ext_vector_type(8)))  float  v8f;

__device__ __forceinline__ bf16 f2bf(float f) {
  union { float f; unsigned u; } in; in.f = f;
  unsigned r = in.u + 0x7FFFu + ((in.u >> 16) & 1u);   // round-to-nearest-even
  union { unsigned short s; bf16 b; } out; out.s = (unsigned short)(r >> 16);
  return out.b;
}

__device__ __forceinline__ v8f wmma_bf16(v16bf a, v16bf b, v8f c) {
  return __builtin_amdgcn_wmma_f32_16x16x32_bf16(false, a, false, b, (short)0, c,
                                                 false, false);
}

// A fragment 16x32 (MxK): lane = M%16; lanes 0-15 hold K 0-15 contiguous,
// lanes 16-31 hold K 16-31 (matches load_b_frag k order).
__device__ __forceinline__ v16bf load_a_frag(const bf16* __restrict__ src, int ld) {
  int lane = threadIdx.x & 31;
  const bf16* p = src + (size_t)(lane & 15) * ld + ((lane >> 4) << 4);
  v8bf lo = *(const v8bf*)p;
  v8bf hi = *(const v8bf*)(p + 8);
  return __builtin_shufflevector(lo, hi, 0,1,2,3,4,5,6,7,8,9,10,11,12,13,14,15);
}

// B fragment 32x16 (KxN) loaded from a row-major [N,K] operand (i.e. B = Wt):
// lane = N%16; lanes 0-15 hold K 0-15 contiguous, lanes 16-31 hold K 16-31.
__device__ __forceinline__ v16bf load_b_frag(const bf16* __restrict__ src, int ld) {
  int lane = threadIdx.x & 31;
  const bf16* p = src + (size_t)(lane & 15) * ld + ((lane >> 4) << 4);
  return *(const v16bf*)p;   // 32B contiguous, 32B aligned
}

// ---------------------------------------------------------------- cast fp32->bf16
__global__ void __launch_bounds__(256) cast_bf16_kernel(const float* __restrict__ s,
                                                        bf16* __restrict__ d, int n) {
  int i = (blockIdx.x * blockDim.x + threadIdx.x) * 4;
  if (i < n) {
    float4 f = *(const float4*)(s + i);
    v4bf o;
    o[0] = f2bf(f.x); o[1] = f2bf(f.y); o[2] = f2bf(f.z); o[3] = f2bf(f.w);
    *(v4bf*)(d + i) = o;
  }
}

// ---------------------------------------------------------------- QKV projection
// out = x @ W^T + b ; Q scaled by 1/sqrt(hd) and stored [B,H,S,hd];
// K stored [B,H,S,hd]; V stored transposed [B,H,hd,S].
__global__ void __launch_bounds__(256) qkv_gemm_kernel(
    const bf16* __restrict__ xb, const bf16* __restrict__ wqkv,
    const float* __restrict__ bq, const float* __restrict__ bk,
    const float* __restrict__ bv,
    bf16* __restrict__ Qb, bf16* __restrict__ Kb, bf16* __restrict__ Vb) {
  const int which = blockIdx.z;
  const bf16* W = wqkv + (size_t)which * DD * DD;
  const float* bias = (which == 0) ? bq : (which == 1) ? bk : bv;
  const int wave = threadIdx.x >> 5;
  const int lane = threadIdx.x & 31;
  const int wm = wave & 3, wn = wave >> 2;
  const int m0 = blockIdx.y * 128 + wm * 32;
  const int n0 = blockIdx.x * 128 + wn * 64;

  const v8f vzero = {0.f, 0.f, 0.f, 0.f, 0.f, 0.f, 0.f, 0.f};
  v8f acc[2][4];
#pragma unroll
  for (int mt = 0; mt < 2; ++mt)
#pragma unroll
    for (int nt = 0; nt < 4; ++nt) acc[mt][nt] = vzero;

  for (int k0 = 0; k0 < DD; k0 += 32) {
    v16bf a0 = load_a_frag(xb + (size_t)m0 * DD + k0, DD);
    v16bf a1 = load_a_frag(xb + (size_t)(m0 + 16) * DD + k0, DD);
    v16bf b0 = load_b_frag(W + (size_t)(n0 +  0) * DD + k0, DD);
    v16bf b1 = load_b_frag(W + (size_t)(n0 + 16) * DD + k0, DD);
    v16bf b2 = load_b_frag(W + (size_t)(n0 + 32) * DD + k0, DD);
    v16bf b3 = load_b_frag(W + (size_t)(n0 + 48) * DD + k0, DD);
    acc[0][0] = wmma_bf16(a0, b0, acc[0][0]);
    acc[0][1] = wmma_bf16(a0, b1, acc[0][1]);
    acc[0][2] = wmma_bf16(a0, b2, acc[0][2]);
    acc[0][3] = wmma_bf16(a0, b3, acc[0][3]);
    acc[1][0] = wmma_bf16(a1, b0, acc[1][0]);
    acc[1][1] = wmma_bf16(a1, b1, acc[1][1]);
    acc[1][2] = wmma_bf16(a1, b2, acc[1][2]);
    acc[1][3] = wmma_bf16(a1, b3, acc[1][3]);
  }

  const int half = lane >> 4, col = lane & 15;
#pragma unroll
  for (int mt = 0; mt < 2; ++mt)
#pragma unroll
    for (int nt = 0; nt < 4; ++nt)
#pragma unroll
      for (int i = 0; i < 8; ++i) {
        int m = m0 + mt * 16 + i + 8 * half;     // C layout: rows in VGPR slots
        int n = n0 + nt * 16 + col;              // cols in lanes
        float val = acc[mt][nt][i] + bias[n];
        int bb = m >> 11, ss = m & (SS - 1);
        int hh = n >> 6,  dd = n & (HDIM - 1);
        if (which == 0)
          Qb[(((size_t)bb * NH + hh) * SS + ss) * HDIM + dd] = f2bf(val * 0.125f);
        else if (which == 1)
          Kb[(((size_t)bb * NH + hh) * SS + ss) * HDIM + dd] = f2bf(val);
        else
          Vb[(((size_t)bb * NH + hh) * HDIM + dd) * SS + ss] = f2bf(val); // V^T
      }
}

// ---------------------------------------------------------------- flash attention
__global__ void __launch_bounds__(128) attn_kernel(
    const bf16* __restrict__ Qb, const bf16* __restrict__ Kb,
    const bf16* __restrict__ Vb, bf16* __restrict__ Ob) {
  __shared__ bf16 pshm[4][16 * 32];   // per-wave private P tile (no barriers needed)
  const int wave = threadIdx.x >> 5;
  const int lane = threadIdx.x & 31;
  const int h = blockIdx.y, b = blockIdx.z;
  const int q0 = blockIdx.x * 64 + wave * 16;
  const bf16* Qh = Qb + (((size_t)b * NH + h) * SS) * HDIM;
  const bf16* Kh = Kb + (((size_t)b * NH + h) * SS) * HDIM;
  const bf16* Vh = Vb + (((size_t)b * NH + h) * HDIM) * SS;

  v16bf aq0 = load_a_frag(Qh + (size_t)q0 * HDIM, HDIM);        // Q 16x64 (scaled)
  v16bf aq1 = load_a_frag(Qh + (size_t)q0 * HDIM + 32, HDIM);

  const v8f vzero = {0.f, 0.f, 0.f, 0.f, 0.f, 0.f, 0.f, 0.f};
  v8f o[4];
#pragma unroll
  for (int nt = 0; nt < 4; ++nt) o[nt] = vzero;
  float rmax[8], rsum[8];
#pragma unroll
  for (int i = 0; i < 8; ++i) { rmax[i] = -3.0e38f; rsum[i] = 0.f; }

  const int half = lane >> 4, col = lane & 15;
  const int kend = q0 + 16;                       // causal: keys 0 .. q0+15
  for (int kt = 0; kt < kend; kt += 32) {
    // scores S = Q K^T : two 16x16 col tiles, K-dim = 64 (2 wmma steps each)
    v8f c0 = vzero, c1 = vzero;
    {
      v16bf bk00 = load_b_frag(Kh + (size_t)(kt)*HDIM, HDIM);
      v16bf bk01 = load_b_frag(Kh + (size_t)(kt)*HDIM + 32, HDIM);
      c0 = wmma_bf16(aq0, bk00, c0);
      c0 = wmma_bf16(aq1, bk01, c0);
      v16bf bk10 = load_b_frag(Kh + (size_t)(kt + 16) * HDIM, HDIM);
      v16bf bk11 = load_b_frag(Kh + (size_t)(kt + 16) * HDIM + 32, HDIM);
      c1 = wmma_bf16(aq0, bk10, c1);
      c1 = wmma_bf16(aq1, bk11, c1);
    }
    if (kt + 32 > q0) {                           // boundary tile: causal mask
#pragma unroll
      for (int i = 0; i < 8; ++i) {
        int row = q0 + i + 8 * half;
        if (kt + col > row)      c0[i] = -3.0e38f;
        if (kt + 16 + col > row) c1[i] = -3.0e38f;
      }
    }
    // online softmax (fp32); row stats reduced across the 16-lane C-layout groups
#pragma unroll
    for (int i = 0; i < 8; ++i) {
      float mcur = fmaxf(c0[i], c1[i]);
#pragma unroll
      for (int off = 1; off < 16; off <<= 1)
        mcur = fmaxf(mcur, __shfl_xor(mcur, off, 32));
      float mnew  = fmaxf(rmax[i], mcur);
      float alpha = __expf(rmax[i] - mnew);
      rmax[i] = mnew;
      float p0 = __expf(c0[i] - mnew);
      float p1 = __expf(c1[i] - mnew);
      float ps = p0 + p1;
#pragma unroll
      for (int off = 1; off < 16; off <<= 1) ps += __shfl_xor(ps, off, 32);
      rsum[i] = rsum[i] * alpha + ps;
#pragma unroll
      for (int nt = 0; nt < 4; ++nt) o[nt][i] *= alpha;
      c0[i] = p0; c1[i] = p1;
    }
    // C-layout -> A-fragment via per-wave LDS round trip
    bf16* pw = &pshm[wave][0];
#pragma unroll
    for (int i = 0; i < 8; ++i) {
      int row = i + 8 * half;
      pw[row * 32 + col]      = f2bf(c0[i]);
      pw[row * 32 + 16 + col] = f2bf(c1[i]);
    }
    const bf16* pp = &pshm[wave][(lane & 15) * 32 + ((lane >> 4) << 4)];
    v8bf plo = *(const v8bf*)pp;
    v8bf phi = *(const v8bf*)(pp + 8);
    v16bf pa = __builtin_shufflevector(plo, phi,
                 0,1,2,3,4,5,6,7,8,9,10,11,12,13,14,15);
    // O += P @ V  (V stored transposed: contiguous-in-key per lane)
#pragma unroll
    for (int nt = 0; nt < 4; ++nt) {
      v16bf bvf = load_b_frag(Vh + (size_t)(nt * 16) * SS + kt, SS);
      o[nt] = wmma_bf16(pa, bvf, o[nt]);
    }
  }
  // epilogue: normalize and store merged heads [B,S,D] as bf16 for out proj
#pragma unroll
  for (int nt = 0; nt < 4; ++nt)
#pragma unroll
    for (int i = 0; i < 8; ++i) {
      int row = q0 + i + 8 * half;
      int dd = nt * 16 + col;
      float val = o[nt][i] / rsum[i];
      Ob[((size_t)b * SS + row) * DD + h * HDIM + dd] = f2bf(val);
    }
}

// ---------------------------------------------------------------- output projection
__global__ void __launch_bounds__(256) out_gemm_kernel(
    const bf16* __restrict__ Ob, const bf16* __restrict__ Wo,
    const float* __restrict__ bo, float* __restrict__ out) {
  const int wave = threadIdx.x >> 5;
  const int lane = threadIdx.x & 31;
  const int wm = wave & 3, wn = wave >> 2;
  const int m0 = blockIdx.y * 128 + wm * 32;
  const int n0 = blockIdx.x * 128 + wn * 64;

  const v8f vzero = {0.f, 0.f, 0.f, 0.f, 0.f, 0.f, 0.f, 0.f};
  v8f acc[2][4];
#pragma unroll
  for (int mt = 0; mt < 2; ++mt)
#pragma unroll
    for (int nt = 0; nt < 4; ++nt) acc[mt][nt] = vzero;

  for (int k0 = 0; k0 < DD; k0 += 32) {
    v16bf a0 = load_a_frag(Ob + (size_t)m0 * DD + k0, DD);
    v16bf a1 = load_a_frag(Ob + (size_t)(m0 + 16) * DD + k0, DD);
    v16bf b0 = load_b_frag(Wo + (size_t)(n0 +  0) * DD + k0, DD);
    v16bf b1 = load_b_frag(Wo + (size_t)(n0 + 16) * DD + k0, DD);
    v16bf b2 = load_b_frag(Wo + (size_t)(n0 + 32) * DD + k0, DD);
    v16bf b3 = load_b_frag(Wo + (size_t)(n0 + 48) * DD + k0, DD);
    acc[0][0] = wmma_bf16(a0, b0, acc[0][0]);
    acc[0][1] = wmma_bf16(a0, b1, acc[0][1]);
    acc[0][2] = wmma_bf16(a0, b2, acc[0][2]);
    acc[0][3] = wmma_bf16(a0, b3, acc[0][3]);
    acc[1][0] = wmma_bf16(a1, b0, acc[1][0]);
    acc[1][1] = wmma_bf16(a1, b1, acc[1][1]);
    acc[1][2] = wmma_bf16(a1, b2, acc[1][2]);
    acc[1][3] = wmma_bf16(a1, b3, acc[1][3]);
  }

  const int half = lane >> 4, col = lane & 15;
#pragma unroll
  for (int mt = 0; mt < 2; ++mt)
#pragma unroll
    for (int nt = 0; nt < 4; ++nt)
#pragma unroll
      for (int i = 0; i < 8; ++i) {
        int m = m0 + mt * 16 + i + 8 * half;
        int n = n0 + nt * 16 + col;
        out[(size_t)m * DD + n] = acc[mt][nt][i] + bo[n];
      }
}

// ---------------------------------------------------------------- launch
extern "C" void kernel_launch(void* const* d_in, const int* in_sizes, int n_in,
                              void* d_out, int out_size, void* d_ws, size_t ws_size,
                              hipStream_t stream) {
  const float* x  = (const float*)d_in[0];
  const float* Wq = (const float*)d_in[1];
  const float* bq = (const float*)d_in[2];
  const float* Wk = (const float*)d_in[3];
  const float* bk = (const float*)d_in[4];
  const float* Wv = (const float*)d_in[5];
  const float* bv = (const float*)d_in[6];
  const float* Wo = (const float*)d_in[7];
  const float* bo = (const float*)d_in[8];
  float* out = (float*)d_out;

  const size_t XE = (size_t)MTOT * DD;  // 3,145,728 elems (x / Q / K / V / O)
  const size_t WE = (size_t)DD * DD;    //   589,824 elems per weight
  bf16* xb   = (bf16*)d_ws;
  bf16* wqkv = xb + XE;                 // Wq,Wk,Wv bf16 back-to-back
  bf16* wo   = wqkv + 3 * WE;
  bf16* Qb   = wo + WE;                 // [B,H,S,hd], pre-scaled by 1/8
  bf16* Kb   = Qb + XE;                 // [B,H,S,hd]
  bf16* Vb   = Kb + XE;                 // [B,H,hd,S]  (transposed)
  bf16* Ob   = Vb + XE;                 // [B,S,D] merged heads

  const int T = 256;
  cast_bf16_kernel<<<dim3((unsigned)((XE / 4 + T - 1) / T)), T, 0, stream>>>(x,  xb,          (int)XE);
  cast_bf16_kernel<<<dim3((unsigned)((WE / 4 + T - 1) / T)), T, 0, stream>>>(Wq, wqkv,        (int)WE);
  cast_bf16_kernel<<<dim3((unsigned)((WE / 4 + T - 1) / T)), T, 0, stream>>>(Wk, wqkv + WE,   (int)WE);
  cast_bf16_kernel<<<dim3((unsigned)((WE / 4 + T - 1) / T)), T, 0, stream>>>(Wv, wqkv + 2*WE, (int)WE);
  cast_bf16_kernel<<<dim3((unsigned)((WE / 4 + T - 1) / T)), T, 0, stream>>>(Wo, wo,          (int)WE);

  qkv_gemm_kernel<<<dim3(DD / 128, MTOT / 128, 3), 256, 0, stream>>>(
      xb, wqkv, bq, bk, bv, Qb, Kb, Vb);

  attn_kernel<<<dim3(SS / 64, NH, NB), 128, 0, stream>>>(Qb, Kb, Vb, Ob);

  out_gemm_kernel<<<dim3(DD / 128, MTOT / 128), 256, 0, stream>>>(Ob, wo, bo, out);
}
